// VectorQuantizer_3762391351926
// MI455X (gfx1250) — matrix operand order repaired and checked by reading, not compiled, not run on hardware
//
#include <hip/hip_runtime.h>

typedef __attribute__((ext_vector_type(16))) __bf16 v16bf;
typedef __attribute__((ext_vector_type(8)))  __bf16 v8bf;
typedef __attribute__((ext_vector_type(8)))  float  v8f;
typedef __attribute__((ext_vector_type(4)))  float  v4f;

#define D_DIM 256
#define K_DIM 1024
#define N_ROWS (32*4096)
#define ROWS_PER_WG 64
#define DSTEPS 8            // D / 32
#define KTILES 64           // K / 16
#define WAVES 8
#define KT_PER_WAVE 8

// workspace layout (bytes)
#define WS_BHI 0                    // 64*8*32*16 bf16 = 512 KB  (B fragments, hi)
#define WS_BLO (512u*1024u)         // 512 KB                    (B fragments, lo)
#define WS_ET  (1024u*1024u)        // 1 MB   embed^T [K][D] f32 (coalesced gather)
#define WS_EN  (2048u*1024u)        // 4 KB   ||e_k||^2

// ---------------------------------------------------------------------------
// Prepass 1: pack embed [D,K] f32 into WMMA B-fragment bf16 hi/lo layout.
// B (32x16, 16-bit) in contiguous k order, same as A: lanes 0-15 hold K=0..15 of the 32-chunk for col
// N=lane, lanes 16-31 hold K=16..31; 16 bf16 per lane packed sequentially.
// grid: 64 x 256  (one thread per (ktile,dstep,lane) fragment lane)
__global__ void k_pack_embed(const float* __restrict__ embed, void* ws) {
  __bf16* bhi = (__bf16*)((char*)ws + WS_BHI);
  __bf16* blo = (__bf16*)((char*)ws + WS_BLO);
  int g     = blockIdx.x * blockDim.x + threadIdx.x;   // 0..16383
  int lane  = g & 31;
  int dstep = (g >> 5) & 7;
  int ktile = g >> 8;
  int dbase = dstep * 32 + ((lane & 16) ? 16 : 0);
  int col   = ktile * 16 + (lane & 15);
#pragma unroll
  for (int j = 0; j < 16; ++j) {
    float  x = embed[(size_t)(dbase + j) * K_DIM + col];
    __bf16 h = (__bf16)x;
    __bf16 l = (__bf16)(x - (float)h);
    bhi[(size_t)g * 16 + j] = h;
    blo[(size_t)g * 16 + j] = l;
  }
}

// Prepass 2: embed^T [K][D] f32 for coalesced code-vector gathers. grid 1024x256
__global__ void k_transpose_embed(const float* __restrict__ embed, void* ws) {
  float* et = (float*)((char*)ws + WS_ET);
  int g = blockIdx.x * blockDim.x + threadIdx.x;       // 0..262143
  int k = g >> 8, d = g & 255;
  et[(size_t)k * D_DIM + d] = embed[(size_t)d * K_DIM + k];
}

// Prepass 3: ||e_k||^2. grid 4x256
__global__ void k_enorm(const float* __restrict__ embed, void* ws) {
  float* en = (float*)((char*)ws + WS_EN);
  int k = blockIdx.x * blockDim.x + threadIdx.x;
  float s = 0.f;
  for (int d = 0; d < D_DIM; ++d) {
    float v = embed[(size_t)d * K_DIM + k];
    s = __builtin_fmaf(v, v, s);
  }
  en[k] = s;
}

// ---------------------------------------------------------------------------
// Main kernel: 64 rows per WG, 8 waves. Wave w owns ktiles [w*8, w*8+8) and all
// 4 row-subtiles (reuses each B fragment 4x). 3x-bf16 split-precision WMMA.
// Register budget: C=32, bestV=32, ttPack=4, B=16, A(transient)=16 -> no spills.
__global__ void __launch_bounds__(256)
k_vq_main(const float* __restrict__ inputs, const void* __restrict__ ws,
          float* __restrict__ out) {
  const __bf16* bhi = (const __bf16*)((const char*)ws + WS_BHI);
  const __bf16* blo = (const __bf16*)((const char*)ws + WS_BLO);
  const float*  et  = (const float*)((const char*)ws + WS_ET);
  const float*  en  = (const float*)((const char*)ws + WS_EN);

  __shared__ __bf16 xhi[ROWS_PER_WG * D_DIM];   // 32 KB
  __shared__ __bf16 xlo[ROWS_PER_WG * D_DIM];   // 32 KB
  __shared__ float  redv[WAVES * ROWS_PER_WG];
  __shared__ int    redi[WAVES * ROWS_PER_WG];
  __shared__ int    fidx[ROWS_PER_WG];

  const int    tid     = threadIdx.x;
  const size_t rowBase = (size_t)blockIdx.x * ROWS_PER_WG;

  // ---- stage x tile -> LDS as bf16 hi/lo (split precision) ----
  const v4f* src = (const v4f*)(inputs + rowBase * D_DIM);
#pragma unroll 4
  for (int it = 0; it < 16; ++it) {
    int e4 = tid + it * 256;
    v4f x  = src[e4];
    int b  = e4 * 4;
#pragma unroll
    for (int j = 0; j < 4; ++j) {
      float  f = x[j];
      __bf16 h = (__bf16)f;
      xhi[b + j] = h;
      xlo[b + j] = (__bf16)(f - (float)h);
    }
  }
  __syncthreads();

  const int wave    = tid >> 5;
  const int lane    = tid & 31;
  const int halfSel = (lane & 16) ? 1 : 0;
  const int koff    = halfSel ? 16 : 0;  // A layout: lanes>=16 hold K=16..31
  const int nidx    = lane & 15;

  float bestV[4][8];   // per (rowsub, C-vgpr) running min of ||e||^2 - 2*dot
  int   ttPack[4];     // 8 x 4-bit fields: winning tt (0..7) per C-vgpr
#pragma unroll
  for (int rs = 0; rs < 4; ++rs) {
    ttPack[rs] = 0;
#pragma unroll
    for (int rr = 0; rr < 8; ++rr) bestV[rs][rr] = 3.4e38f;
  }

#pragma unroll 1
  for (int tt = 0; tt < KT_PER_WAVE; ++tt) {
    const int t = wave * KT_PER_WAVE + tt;
    v8f C[4];
#pragma unroll
    for (int rs = 0; rs < 4; ++rs) C[rs] = v8f{};

#pragma unroll 1
    for (int s = 0; s < DSTEPS; ++s) {
      size_t fb = ((size_t)(t * DSTEPS + s) * 32 + lane) * 16;
      v16bf Bh = *(const v16bf*)(bhi + fb);
      v16bf Bl = *(const v16bf*)(blo + fb);
#pragma unroll
      for (int rs = 0; rs < 4; ++rs) {
        int  row = rs * 16 + nidx;
        int  b0  = row * D_DIM + s * 32 + koff;   // K chunk: [koff..koff+15]
        v8bf ah0 = *(const v8bf*)(xhi + b0);
        v8bf ah1 = *(const v8bf*)(xhi + b0 + 8);
        v8bf al0 = *(const v8bf*)(xlo + b0);
        v8bf al1 = *(const v8bf*)(xlo + b0 + 8);
        v16bf Ah = __builtin_shufflevector(ah0, ah1, 0,1,2,3,4,5,6,7,8,9,10,11,12,13,14,15);
        v16bf Al = __builtin_shufflevector(al0, al1, 0,1,2,3,4,5,6,7,8,9,10,11,12,13,14,15);
        C[rs] = __builtin_amdgcn_wmma_f32_16x16x32_bf16(false, Ah, false, Bh, (short)0, C[rs], false, false);
        C[rs] = __builtin_amdgcn_wmma_f32_16x16x32_bf16(false, Ah, false, Bl, (short)0, C[rs], false, false);
        C[rs] = __builtin_amdgcn_wmma_f32_16x16x32_bf16(false, Al, false, Bh, (short)0, C[rs], false, false);
      }
    }

    // distance surrogate: ||e||^2 - 2*dot  (||x||^2 constant per row -> dropped).
    // ci strictly increases with tt within a lane -> strict '<' keeps first index.
    float ev = en[t * 16 + nidx];
#pragma unroll
    for (int rs = 0; rs < 4; ++rs) {
#pragma unroll
      for (int rr = 0; rr < 8; ++rr) {
        float sv  = __builtin_fmaf(-2.f, C[rs][rr], ev);
        bool  win = sv < bestV[rs][rr];
        int   ins = (ttPack[rs] & ~(0xF << (rr * 4))) | (tt << (rr * 4));
        bestV[rs][rr] = win ? sv  : bestV[rs][rr];
        ttPack[rs]    = win ? ins : ttPack[rs];
      }
    }
  }

  // ---- argmin across the 16 lanes of each half (C: N = lane&15) ----
#pragma unroll
  for (int rs = 0; rs < 4; ++rs)
#pragma unroll
    for (int rr = 0; rr < 8; ++rr) {
      float v  = bestV[rs][rr];
      int   wt = (ttPack[rs] >> (rr * 4)) & 0xF;
      int   i  = (wave * KT_PER_WAVE + wt) * 16 + nidx;   // winning column
#pragma unroll
      for (int mask = 1; mask <= 8; mask <<= 1) {
        float ov = __shfl_xor(v, mask, 32);
        int   oi = __shfl_xor(i, mask, 32);
        if (ov < v || (ov == v && oi < i)) { v = ov; i = oi; }
      }
      int rowLocal = rs * 16 + rr + halfSel * 8;   // lanes>=16 carry M = rr+8
      if (nidx == 0) {
        redv[wave * ROWS_PER_WG + rowLocal] = v;
        redi[wave * ROWS_PER_WG + rowLocal] = i;
      }
    }
  __syncthreads();

  // ---- cross-wave reduce (waves ascend in ktile -> first-index ties OK) ----
  if (tid < ROWS_PER_WG) {
    float bv = redv[tid];
    int   bi = redi[tid];
#pragma unroll
    for (int w = 1; w < WAVES; ++w) {
      float v = redv[w * ROWS_PER_WG + tid];
      int   i = redi[w * ROWS_PER_WG + tid];
      if (v < bv || (v == bv && i < bi)) { bv = v; bi = i; }
    }
    fidx[tid] = bi;
    out[(size_t)3 * N_ROWS * D_DIM + rowBase + tid] = (float)bi;  // indices
  }
  __syncthreads();

  // ---- gather code vectors + emit quantized_st and codes (coalesced) ----
  float* qout  = out;
  float* codes = out + (size_t)N_ROWS * D_DIM;
#pragma unroll 1
  for (int r = 0; r < ROWS_PER_WG; ++r) {
    size_t rg  = rowBase + r;
    int    idx = fidx[r];
    int    d   = tid;
    float  q   = et[(size_t)idx * D_DIM + d];          // contiguous row of E^T
    float  xin = inputs[rg * D_DIM + d];
    qout[rg * D_DIM + d]                = q;           // quantized_st == quantized
    codes[rg * (2 * D_DIM) + d]         = xin;
    codes[rg * (2 * D_DIM) + D_DIM + d] = q;
  }
}

// ---------------------------------------------------------------------------
extern "C" void kernel_launch(void* const* d_in, const int* in_sizes, int n_in,
                              void* d_out, int out_size, void* d_ws, size_t ws_size,
                              hipStream_t stream) {
  const float* inputs = (const float*)d_in[0];   // [32,4096,256] f32
  const float* embed  = (const float*)d_in[1];   // [256,1024]    f32
  float* out = (float*)d_out;
  (void)in_sizes; (void)n_in; (void)out_size; (void)ws_size;

  k_pack_embed     <<<64,   256, 0, stream>>>(embed, d_ws);
  k_transpose_embed<<<1024, 256, 0, stream>>>(embed, d_ws);
  k_enorm          <<<4,    256, 0, stream>>>(embed, d_ws);
  k_vq_main<<<N_ROWS / ROWS_PER_WG, 256, 0, stream>>>(inputs, d_ws, out);
}
